// GraphConv_35373350650219
// MI455X (gfx1250) — matrix operand order repaired and checked by reading, not compiled, not run on hardware
//
#include <hip/hip_runtime.h>
#include <hip/hip_bf16.h>

#define HDIM    128
#define EDIM    32
#define KMSG    160      // HDIM + EDIM
#define NNODES  100000
#define NEDGES  1600000

typedef __attribute__((ext_vector_type(8)))  __bf16 v8bf;
typedef __attribute__((ext_vector_type(16))) __bf16 v16bf;
typedef __attribute__((ext_vector_type(8)))  float  v8f;

__device__ __forceinline__ v16bf cat16(v8bf lo, v8bf hi) {
  return __builtin_shufflevector(lo, hi, 0,1,2,3,4,5,6,7,8,9,10,11,12,13,14,15);
}

// Build one A-fragment (v16bf) for this lane directly from a global f32 row.
// Per the ISA 16-bit A layout: halfs[0..7] = row[o .. o+7], halfs[8..15] = row[o+16 .. o+23].
__device__ __forceinline__ v16bf ldA(const float* __restrict__ p, int o) {
  const float4 f0 = *(const float4*)(p + o);
  const float4 f1 = *(const float4*)(p + o + 4);
  const float4 f2 = *(const float4*)(p + o + 16);
  const float4 f3 = *(const float4*)(p + o + 20);
  v16bf a;
  a[0]  = (__bf16)f0.x; a[1]  = (__bf16)f0.y; a[2]  = (__bf16)f0.z; a[3]  = (__bf16)f0.w;
  a[4]  = (__bf16)f1.x; a[5]  = (__bf16)f1.y; a[6]  = (__bf16)f1.z; a[7]  = (__bf16)f1.w;
  a[8]  = (__bf16)f2.x; a[9]  = (__bf16)f2.y; a[10] = (__bf16)f2.z; a[11] = (__bf16)f2.w;
  a[12] = (__bf16)f3.x; a[13] = (__bf16)f3.y; a[14] = (__bf16)f3.z; a[15] = (__bf16)f3.w;
  return a;
}

__device__ __forceinline__ int bperm(int srcLane, int v) {
  return __builtin_amdgcn_ds_bpermute(srcLane << 2, v);
}

// ---------------------------------------------------------------------------
// Node kernel: out[n, :] = x[n, :] @ W_self + b_self   (pre-activation, f32)
// LDS: Bt[128][136] bf16 (34816 B) only.
// ---------------------------------------------------------------------------
#define NB_STRIDE 136

__global__ void __launch_bounds__(128)
gc_node_kernel(const float* __restrict__ x, const float* __restrict__ Wself,
               const float* __restrict__ bself, float* __restrict__ out)
{
  extern __shared__ char smem[];
  __bf16* ldsB = (__bf16*)smem;   // 34816 B

  const int tid  = threadIdx.x;
  const int lane = tid & 31;
  const int wave = tid >> 5;

  // Stage W_self^T as bf16: ldsB[n*136 + k] = W_self[k][n]
  {
    const int n = tid;            // blockDim.x == 128
    #pragma unroll 4
    for (int k = 0; k < HDIM; ++k)
      ldsB[n * NB_STRIDE + k] = (__bf16)Wself[k * HDIM + n];
  }
  __syncthreads();

  const int m     = lane & 15;
  const int part  = lane >> 4;
  const int mbase = part * 8;    // C/D row group for this lane
  const int nlane = lane & 15;
  const int kb_a  = part * 8;    // A-fragment K sub-offset (ISA layout)
  const int kb_b  = part * 8;    // B-fragment K sub-offset (ISA layout)

  float bvals[8];                // loop-invariant bias per N-tile
  #pragma unroll
  for (int nt = 0; nt < 8; ++nt) bvals[nt] = bself[nt * 16 + nlane];

  const int nTiles  = NNODES / 16;
  const int wgid    = blockIdx.x * 4 + wave;
  const int wstride = gridDim.x * 4;

  for (int tile = wgid; tile < nTiles; tile += wstride) {
    const int n0 = tile * 16;
    const float* __restrict__ xr = x + (size_t)(n0 + m) * HDIM;

    v8f acc[8] = {};   // zero C -> first WMMA uses inline 0 operand
    #pragma unroll
    for (int kc = 0; kc < 4; ++kc) {
      const v16bf a = ldA(xr, kc * 32 + kb_a);
      #pragma unroll
      for (int nt = 0; nt < 8; ++nt) {
        const __bf16* pb = ldsB + (nt * 16 + nlane) * NB_STRIDE + kc * 32 + kb_b;
        const v16bf b = cat16(*(const v8bf*)pb, *(const v8bf*)(pb + 16));
        acc[nt] = __builtin_amdgcn_wmma_f32_16x16x32_bf16(
            false, a, false, b, (short)0, acc[nt], false, false);
      }
    }

    #pragma unroll
    for (int nt = 0; nt < 8; ++nt) {
      const int ncol = nt * 16 + nlane;
      #pragma unroll
      for (int r = 0; r < 8; ++r)
        out[(size_t)(n0 + mbase + r) * HDIM + ncol] = acc[nt][r] + bvals[nt];
    }
  }
}

// ---------------------------------------------------------------------------
// Edge kernel: m = [x[src] | edge_attr] @ W_msg + b_msg ; atomic-add into out[dst]
// Each wave processes 32 edges (two 16-row M-tiles) per iteration so every
// B fragment read from LDS feeds two WMMAs. A fragments come straight from
// global memory (one read per element), indices broadcast via ds_bpermute.
// LDS: Bt[128][168] bf16 (43008 B) only.
// ---------------------------------------------------------------------------
#define EB_STRIDE 168

__global__ void __launch_bounds__(128)
gc_edge_kernel(const float* __restrict__ x, const int* __restrict__ eidx,
               const float* __restrict__ eattr, const float* __restrict__ Wmsg,
               const float* __restrict__ bmsg, float* __restrict__ out)
{
  extern __shared__ char smem[];
  __bf16* ldsB = (__bf16*)smem;   // 43008 B

  const int tid  = threadIdx.x;
  const int lane = tid & 31;
  const int wave = tid >> 5;

  // Stage W_msg^T as bf16: ldsB[n*168 + k] = W_msg[k][n], k in [0,160)
  {
    const int n = tid;            // blockDim.x == 128
    #pragma unroll 4
    for (int k = 0; k < KMSG; ++k)
      ldsB[n * EB_STRIDE + k] = (__bf16)Wmsg[k * HDIM + n];
  }
  __syncthreads();

  const int m     = lane & 15;
  const int part  = lane >> 4;
  const int mbase = part * 8;
  const int nlane = lane & 15;
  const int kb_a  = part * 8;
  const int kb_b  = part * 8;

  float bvals[8];
  #pragma unroll
  for (int nt = 0; nt < 8; ++nt) bvals[nt] = bmsg[nt * 16 + nlane];

  const int nIter   = NEDGES / 32;         // 32 edges per wave-iteration
  const int wgid    = blockIdx.x * 4 + wave;
  const int wstride = gridDim.x * 4;

  for (int it = wgid; it < nIter; it += wstride) {
    const int e0 = it * 32;

    // Coalesced index loads: lane j holds edge e0+j.
    const int sA = eidx[e0 + lane];            // src
    const int dA = eidx[NEDGES + e0 + lane];   // dst

    // Prefetch next iteration's index block (global_prefetch_b8).
    __builtin_prefetch(eidx + e0 + wstride * 32 + lane, 0, 1);

    // Broadcast the src node for the A-row this lane owns in each M-tile.
    const int s0 = bperm(m,      sA);
    const int s1 = bperm(16 + m, sA);
    const float* __restrict__ r0 = x + (size_t)s0 * HDIM;
    const float* __restrict__ r1 = x + (size_t)s1 * HDIM;
    const float* __restrict__ ea0 = eattr + (size_t)(e0 + m) * EDIM;
    const float* __restrict__ ea1 = eattr + (size_t)(e0 + 16 + m) * EDIM;

    v8f acc[16] = {};   // [0..7] = tile0 N-tiles, [8..15] = tile1 N-tiles

    #pragma unroll
    for (int kc = 0; kc < 5; ++kc) {           // K = 160 = 5 x 32
      const v16bf a0 = (kc < 4) ? ldA(r0, kc * 32 + kb_a) : ldA(ea0, kb_a);
      const v16bf a1 = (kc < 4) ? ldA(r1, kc * 32 + kb_a) : ldA(ea1, kb_a);
      #pragma unroll
      for (int nt = 0; nt < 8; ++nt) {
        const __bf16* pb = ldsB + (nt * 16 + nlane) * EB_STRIDE + kc * 32 + kb_b;
        const v16bf b = cat16(*(const v8bf*)pb, *(const v8bf*)(pb + 16));
        acc[nt]     = __builtin_amdgcn_wmma_f32_16x16x32_bf16(
            false, a0, false, b, (short)0, acc[nt],     false, false);
        acc[8 + nt] = __builtin_amdgcn_wmma_f32_16x16x32_bf16(
            false, a1, false, b, (short)0, acc[8 + nt], false, false);
      }
    }

    // Scatter-add (segment_sum): L2-resident fire-and-forget f32 atomics.
    #pragma unroll
    for (int t = 0; t < 2; ++t) {
      int dsts[8];
      #pragma unroll
      for (int r = 0; r < 8; ++r) dsts[r] = bperm(t * 16 + mbase + r, dA);
      #pragma unroll
      for (int nt = 0; nt < 8; ++nt) {
        const int ncol = nt * 16 + nlane;
        #pragma unroll
        for (int r = 0; r < 8; ++r)
          atomicAdd(out + (size_t)dsts[r] * HDIM + ncol,
                    acc[t * 8 + nt][r] + bvals[nt]);
      }
    }
  }
}

// ---------------------------------------------------------------------------
// ReLU over the accumulated output
// ---------------------------------------------------------------------------
__global__ void __launch_bounds__(256)
gc_relu_kernel(float* __restrict__ out, int n4)
{
  int i = blockIdx.x * blockDim.x + threadIdx.x;
  const int stride = gridDim.x * blockDim.x;
  float4* p = (float4*)out;
  for (; i < n4; i += stride) {
    float4 v = p[i];
    v.x = fmaxf(v.x, 0.0f); v.y = fmaxf(v.y, 0.0f);
    v.z = fmaxf(v.z, 0.0f); v.w = fmaxf(v.w, 0.0f);
    p[i] = v;
  }
}

extern "C" void kernel_launch(void* const* d_in, const int* in_sizes, int n_in,
                              void* d_out, int out_size, void* d_ws, size_t ws_size,
                              hipStream_t stream) {
  (void)in_sizes; (void)n_in; (void)out_size; (void)d_ws; (void)ws_size;
  const float* x  = (const float*)d_in[0];
  const int*   ei = (const int*)d_in[1];     // [2, E] int32: src row 0, dst row 1
  const float* ea = (const float*)d_in[2];
  const float* Ws = (const float*)d_in[3];
  const float* bs = (const float*)d_in[4];
  const float* Wm = (const float*)d_in[5];
  const float* bm = (const float*)d_in[6];
  float* out = (float*)d_out;

  const size_t shNode = 34816;   // W_self^T bf16
  const size_t shEdge = 43008;   // W_msg^T bf16

  gc_node_kernel<<<1024, 128, shNode, stream>>>(x, Ws, bs, out);
  gc_edge_kernel<<<2048, 128, shEdge, stream>>>(x, ei, ea, Wm, bm, out);
  gc_relu_kernel<<<4096, 256, 0, stream>>>(out, (NNODES * HDIM) / 4);
}
